// WindowedTransformerEncoder_31619549233184
// MI455X (gfx1250) — matrix operand order repaired and checked by reading, not compiled, not run on hardware
//
#include <hip/hip_runtime.h>
#include <hip/hip_bf16.h>

typedef _Float16 h8  __attribute__((ext_vector_type(8)));
typedef _Float16 h16 __attribute__((ext_vector_type(16)));
typedef float    f8  __attribute__((ext_vector_type(8)));

#define WMMA_F16(a, b, c) \
  __builtin_amdgcn_wmma_f32_16x16x32_f16(false, (a), false, (b), (short)0, (c), false, false)

#if __has_builtin(__builtin_amdgcn_global_load_async_to_lds_b128)
#define HAVE_ASYNC_LDS 1
typedef int v4i_t __attribute__((ext_vector_type(4)));
typedef __attribute__((address_space(1))) void*   as1_vp;
typedef __attribute__((address_space(3))) void*   as3_vp;
typedef __attribute__((address_space(1))) v4i_t*  as1_v4p;
typedef __attribute__((address_space(3))) v4i_t*  as3_v4p;
__device__ __forceinline__ void async_cp16B(const _Float16* g, _Float16* l) {
  as1_vp gv = (as1_vp)(void*)const_cast<_Float16*>(g);  // addrspacecast only
  as3_vp lv = (as3_vp)(void*)l;
  __builtin_amdgcn_global_load_async_to_lds_b128((as1_v4p)gv, (as3_v4p)lv, 0, 0);
}
__device__ __forceinline__ void wait_async0() {
#if __has_builtin(__builtin_amdgcn_s_wait_asynccnt)
  __builtin_amdgcn_s_wait_asynccnt(0);
#else
  asm volatile("s_wait_asynccnt 0x0" ::: "memory");
#endif
}
#endif

// ---------------------------------------------------------------------------
// Fragment loaders (CDNA5 wave32 WMMA layouts, CDNA5 ISA)
// A 16x32 f16: lanes 0-15 -> M=lane, K={0..7,16..23}; lanes 16-31 -> K={8..15,24..31}
// B 32x16 f16 (stored LDS[n][k]): lane -> N=lane&15, K={0..7,16..23} (lanes 0-15), {8..15,24..31} (lanes 16-31), as A
// D 16x16 f32: vgpr r -> M = r + 8*(lane>>4), N = lane&15
// ---------------------------------------------------------------------------
__device__ __forceinline__ h16 ld_afrag(const _Float16* base, int ldk, int row0,
                                        int k0, int lane) {
  int row = row0 + (lane & 15);
  int g   = lane >> 4;
  const _Float16* p = base + row * ldk + k0 + 8 * g;
  h8 lo = *(const h8*)(p);
  h8 hi = *(const h8*)(p + 16);
  return __builtin_shufflevector(lo, hi, 0,1,2,3,4,5,6,7,8,9,10,11,12,13,14,15);
}

__device__ __forceinline__ h16 ld_bfrag(const _Float16* base, int ldk, int n0,
                                        int k0, int lane) {
  int n = n0 + (lane & 15);
  int g = lane >> 4;
  const _Float16* p = base + n * ldk + k0 + 8 * g;
  h8 lo = *(const h8*)(p);
  h8 hi = *(const h8*)(p + 16);
  return __builtin_shufflevector(lo, hi, 0,1,2,3,4,5,6,7,8,9,10,11,12,13,14,15);
}

// ---------------------------------------------------------------------------
// GEMM core: C[M x N] = A[M x K] * Bt[N x K]^T, f16 in, f32 acc.
// 256 threads (8 waves). Block tile 128x128, BK=32. Wave tile 32x64 (2x4 frags).
// Double-buffered LDS; async global->LDS loads (ASYNCcnt) when available.
// blockIdx.x = M tile, blockIdx.y = N tile. N, K multiples of 128/32; M ragged.
// ---------------------------------------------------------------------------
__device__ __forceinline__ void gemm_core(const _Float16* __restrict__ A,
                                          const _Float16* __restrict__ Bt,
                                          int M, int K, f8 (&acc)[2][4]) {
  __shared__ _Float16 As[2][128 * 32];
  __shared__ _Float16 Bs[2][128 * 32];
  const int tid  = threadIdx.x;
  const int lane = tid & 31;
  const int wave = tid >> 5;
  const int wm   = wave >> 1;   // 0..3
  const int wn   = wave & 1;    // 0..1
  const int bm   = blockIdx.x;
  const int bn   = blockIdx.y;

  for (int i = 0; i < 2; ++i)
    for (int j = 0; j < 4; ++j)
      for (int r = 0; r < 8; ++r) acc[i][j][r] = 0.f;

  const int r  = tid >> 1;          // 0..127
  const int c0 = (tid & 1) * 16;    // 0 or 16
  int ga = bm * 128 + r; if (ga > M - 1) ga = M - 1;   // clamp ragged M
  const int gb = bn * 128 + r;
  const _Float16* gA = A + (size_t)ga * K + c0;
  const _Float16* gB = Bt + (size_t)gb * K + c0;
  const int ksteps = K >> 5;

#if defined(HAVE_ASYNC_LDS)
  // ---- async double-buffered pipeline: one barrier per K-step ----
  {
    async_cp16B(gA,     &As[0][r * 32 + c0]);
    async_cp16B(gA + 8, &As[0][r * 32 + c0 + 8]);
    async_cp16B(gB,     &Bs[0][r * 32 + c0]);
    async_cp16B(gB + 8, &Bs[0][r * 32 + c0 + 8]);
  }
  for (int kb = 0; kb < ksteps; ++kb) {
    const int cur = kb & 1;
    wait_async0();          // this wave's async loads for buf[cur] landed
    __syncthreads();        // everyone's landed; buf[cur^1] fully consumed
    if (kb + 1 < ksteps) {  // prefetch next tile into the other buffer
      const _Float16* nA = gA + (size_t)(kb + 1) * 32;
      const _Float16* nB = gB + (size_t)(kb + 1) * 32;
      async_cp16B(nA,     &As[cur ^ 1][r * 32 + c0]);
      async_cp16B(nA + 8, &As[cur ^ 1][r * 32 + c0 + 8]);
      async_cp16B(nB,     &Bs[cur ^ 1][r * 32 + c0]);
      async_cp16B(nB + 8, &Bs[cur ^ 1][r * 32 + c0 + 8]);
    }
    h16 af[2], bf[4];
#pragma unroll
    for (int i = 0; i < 2; ++i)
      af[i] = ld_afrag(&As[cur][0], 32, wm * 32 + i * 16, 0, lane);
#pragma unroll
    for (int j = 0; j < 4; ++j)
      bf[j] = ld_bfrag(&Bs[cur][0], 32, wn * 64 + j * 16, 0, lane);
#pragma unroll
    for (int i = 0; i < 2; ++i)
#pragma unroll
      for (int j = 0; j < 4; ++j) acc[i][j] = WMMA_F16(af[i], bf[j], acc[i][j]);
  }
  __syncthreads();
#else
  // ---- synchronous fallback (VGPR staging + prefetch of next tile) ----
  for (int kb = 0; kb < ksteps; ++kb) {
    const _Float16* sa = gA + (size_t)kb * 32;
    const _Float16* sb = gB + (size_t)kb * 32;
    if (kb + 1 < ksteps) {
      __builtin_prefetch(sa + 32, 0, 1);   // global_prefetch_b8
      __builtin_prefetch(sb + 32, 0, 1);
    }
    *(h8*)(&As[0][r * 32 + c0])     = *(const h8*)(sa);
    *(h8*)(&As[0][r * 32 + c0 + 8]) = *(const h8*)(sa + 8);
    *(h8*)(&Bs[0][r * 32 + c0])     = *(const h8*)(sb);
    *(h8*)(&Bs[0][r * 32 + c0 + 8]) = *(const h8*)(sb + 8);
    __syncthreads();
    h16 af[2], bf[4];
#pragma unroll
    for (int i = 0; i < 2; ++i)
      af[i] = ld_afrag(&As[0][0], 32, wm * 32 + i * 16, 0, lane);
#pragma unroll
    for (int j = 0; j < 4; ++j)
      bf[j] = ld_bfrag(&Bs[0][0], 32, wn * 64 + j * 16, 0, lane);
#pragma unroll
    for (int i = 0; i < 2; ++i)
#pragma unroll
      for (int j = 0; j < 4; ++j) acc[i][j] = WMMA_F16(af[i], bf[j], acc[i][j]);
    __syncthreads();
  }
#endif
}

#define EPILOGUE_COORDS()                                                     \
  const int lane = threadIdx.x & 31;                                          \
  const int wave = threadIdx.x >> 5;                                          \
  const int wm = wave >> 1, wn = wave & 1;                                    \
  const int mbase = blockIdx.x * 128 + wm * 32 + 8 * (lane >> 4);             \
  const int nbase = blockIdx.y * 128 + wn * 64 + (lane & 15);

// ---------------------------------------------------------------------------
// GEMM kernels with epilogues
// ---------------------------------------------------------------------------
__global__ __launch_bounds__(256) void k_gemm_qkv(
    const _Float16* __restrict__ A, const _Float16* __restrict__ Bt,
    const float* __restrict__ bias, _Float16* __restrict__ qo,
    _Float16* __restrict__ ko, _Float16* __restrict__ vo, int M, int K) {
  f8 acc[2][4];
  gemm_core(A, Bt, M, K, acc);
  EPILOGUE_COORDS();
  for (int i = 0; i < 2; ++i)
    for (int j = 0; j < 4; ++j)
      for (int r = 0; r < 8; ++r) {
        int m = mbase + i * 16 + r;
        int n = nbase + j * 16;
        if (m < M) {
          float val = acc[i][j][r] + bias[n];
          int s = n / 768, rem = n - s * 768;
          int hh = rem >> 6, d = rem & 63;
          int bw = m / 196, nt = m - bw * 196;
          size_t dst = ((size_t)(bw * 12 + hh) * 196 + nt) * 64 + d;
          _Float16 hv = (_Float16)val;
          if (s == 0) qo[dst] = hv;
          else if (s == 1) ko[dst] = hv;
          else vo[dst] = hv;
        }
      }
}

__global__ __launch_bounds__(256) void k_gemm_proj(
    const _Float16* __restrict__ A, const _Float16* __restrict__ Bt,
    const float* __restrict__ bias, const float* __restrict__ xin,
    float* __restrict__ out, int M, int K) {
  f8 acc[2][4];
  gemm_core(A, Bt, M, K, acc);
  EPILOGUE_COORDS();
  for (int i = 0; i < 2; ++i)
    for (int j = 0; j < 4; ++j)
      for (int r = 0; r < 8; ++r) {
        int m = mbase + i * 16 + r;
        int n = nbase + j * 16;
        if (m < M) {
          float val = acc[i][j][r] + bias[n];
          int bw = m / 196, nt = m - bw * 196;
          int b = bw / 25, r2 = bw - b * 25;
          int wh = r2 / 5, ww = r2 - wh * 5;
          int i0 = nt / 14, j0 = nt - i0 * 14;
          int hh = wh * 14 + i0, wp = ww * 14 + j0;
          if (hh < 64 && wp < 64) {   // crop: drop padded pixels
            size_t idx = (((size_t)(b * 64 + hh) * 64) + wp) * 768 + n;
            out[idx] = xin[idx] + val;   // x2 = shortcut + proj(o)
          }
        }
      }
}

__global__ __launch_bounds__(256) void k_gemm_gelu(
    const _Float16* __restrict__ A, const _Float16* __restrict__ Bt,
    const float* __restrict__ bias, _Float16* __restrict__ mid, int M, int K) {
  f8 acc[2][4];
  gemm_core(A, Bt, M, K, acc);
  EPILOGUE_COORDS();
  for (int i = 0; i < 2; ++i)
    for (int j = 0; j < 4; ++j)
      for (int r = 0; r < 8; ++r) {
        int m = mbase + i * 16 + r;
        int n = nbase + j * 16;
        if (m < M) {
          float xv = acc[i][j][r] + bias[n];
          float g  = 0.5f * xv * (1.0f + erff(xv * 0.70710678118654752f));
          mid[(size_t)m * 3072 + n] = (_Float16)g;
        }
      }
}

__global__ __launch_bounds__(256) void k_gemm_out(
    const _Float16* __restrict__ A, const _Float16* __restrict__ Bt,
    const float* __restrict__ bias, float* __restrict__ out, int M, int K) {
  f8 acc[2][4];
  gemm_core(A, Bt, M, K, acc);
  EPILOGUE_COORDS();
  for (int i = 0; i < 2; ++i)
    for (int j = 0; j < 4; ++j)
      for (int r = 0; r < 8; ++r) {
        int m = mbase + i * 16 + r;
        int n = nbase + j * 16;
        if (m < M) {
          size_t idx = (size_t)m * 768 + n;
          out[idx] = out[idx] + acc[i][j][r] + bias[n];   // x2 + mlp
        }
      }
}

// ---------------------------------------------------------------------------
// Weight convert fp32 [K][N] -> f16 [N][K] (LDS-tiled transpose)
// ---------------------------------------------------------------------------
__global__ void k_wt(const float* __restrict__ W, _Float16* __restrict__ Wt,
                     int K, int N) {
  __shared__ float tile[32][33];
  int tx = threadIdx.x, ty = threadIdx.y;           // block (32,8)
  int n0 = blockIdx.x * 32, k0 = blockIdx.y * 32;
  for (int dy = 0; dy < 32; dy += 8)
    tile[ty + dy][tx] = W[(size_t)(k0 + ty + dy) * N + n0 + tx];
  __syncthreads();
  for (int dy = 0; dy < 32; dy += 8)
    Wt[(size_t)(n0 + ty + dy) * K + k0 + tx] = (_Float16)tile[tx][ty + dy];
}

// ---------------------------------------------------------------------------
// LayerNorm (C = 768, 256 threads x 3 elems)
// ---------------------------------------------------------------------------
__device__ __forceinline__ float block_sum256(float v, float* red) {
  int tid = threadIdx.x;
  red[tid] = v; __syncthreads();
  for (int s = 128; s > 0; s >>= 1) {
    if (tid < s) red[tid] += red[tid + s];
    __syncthreads();
  }
  float r = red[0]; __syncthreads();
  return r;
}

// LN1 fused with window partition: token t in (Bw=200, N=196) order; zero pad.
__global__ __launch_bounds__(256) void k_ln1(const float* __restrict__ x,
                                             const float* __restrict__ g,
                                             const float* __restrict__ bb,
                                             _Float16* __restrict__ xwin) {
  __shared__ float red[256];
  int t = blockIdx.x, tid = threadIdx.x;
  int bw = t / 196, nt = t - bw * 196;
  int b = bw / 25, r2 = bw - b * 25;
  int wh = r2 / 5, ww = r2 - wh * 5;
  int i0 = nt / 14, j0 = nt - i0 * 14;
  int hh = wh * 14 + i0, wp = ww * 14 + j0;
  _Float16* dst = xwin + (size_t)t * 768;
  if (hh >= 64 || wp >= 64) {                // padded token -> zeros
    for (int e = tid; e < 768; e += 256) dst[e] = (_Float16)0.f;
    return;
  }
  const float* src = x + (((size_t)(b * 64 + hh) * 64) + wp) * 768;
  float v0 = src[tid], v1 = src[tid + 256], v2 = src[tid + 512];
  float mean = block_sum256(v0 + v1 + v2, red) * (1.f / 768.f);
  float d0 = v0 - mean, d1 = v1 - mean, d2 = v2 - mean;
  float var = block_sum256(d0 * d0 + d1 * d1 + d2 * d2, red) * (1.f / 768.f);
  float rstd = rsqrtf(var + 1e-6f);
  dst[tid]       = (_Float16)(d0 * rstd * g[tid]       + bb[tid]);
  dst[tid + 256] = (_Float16)(d1 * rstd * g[tid + 256] + bb[tid + 256]);
  dst[tid + 512] = (_Float16)(d2 * rstd * g[tid + 512] + bb[tid + 512]);
}

__global__ __launch_bounds__(256) void k_ln2(const float* __restrict__ x2,
                                             const float* __restrict__ g,
                                             const float* __restrict__ bb,
                                             _Float16* __restrict__ h2) {
  __shared__ float red[256];
  int t = blockIdx.x, tid = threadIdx.x;
  const float* src = x2 + (size_t)t * 768;
  _Float16* dst = h2 + (size_t)t * 768;
  float v0 = src[tid], v1 = src[tid + 256], v2 = src[tid + 512];
  float mean = block_sum256(v0 + v1 + v2, red) * (1.f / 768.f);
  float d0 = v0 - mean, d1 = v1 - mean, d2 = v2 - mean;
  float var = block_sum256(d0 * d0 + d1 * d1 + d2 * d2, red) * (1.f / 768.f);
  float rstd = rsqrtf(var + 1e-6f);
  dst[tid]       = (_Float16)(d0 * rstd * g[tid]       + bb[tid]);
  dst[tid + 256] = (_Float16)(d1 * rstd * g[tid + 256] + bb[tid + 256]);
  dst[tid + 512] = (_Float16)(d2 * rstd * g[tid + 512] + bb[tid + 512]);
}

// ---------------------------------------------------------------------------
// Attention: one workgroup per (window, head). N=196 pad->208(M)/224(K keys).
// LDS: S[208][224] f32, rel tables, q[208][64], k[224][64], v^T[64][224] f16.
// ---------------------------------------------------------------------------
#define SM_S   0
#define SM_RH  186368
#define SM_RW  193280
#define SM_Q   200192
#define SM_K   226816
#define SM_V   255488
#define SM_TOT 284160

__global__ __launch_bounds__(256) void k_attn(
    const _Float16* __restrict__ qg, const _Float16* __restrict__ kg,
    const _Float16* __restrict__ vg, const float* __restrict__ rel_h,
    const float* __restrict__ rel_w, _Float16* __restrict__ o) {
  extern __shared__ char smem[];
  float*    S  = (float*)(smem + SM_S);
  float*    rh = (float*)(smem + SM_RH);
  float*    rw = (float*)(smem + SM_RW);
  _Float16* qs = (_Float16*)(smem + SM_Q);
  _Float16* ks = (_Float16*)(smem + SM_K);
  _Float16* vt = (_Float16*)(smem + SM_V);

  const int tid = threadIdx.x, lane = tid & 31, wave = tid >> 5;
  const int hw = blockIdx.x;           // = bw*12 + head
  const int bw = hw / 12, head = hw - bw * 12;
  const size_t base = (size_t)hw * 196 * 64;

  // stage q (pad rows to 208 with zeros) and k (pad to 224) as h8 units
  for (int u = tid; u < 208 * 8; u += 256) {
    h8 val;
    if ((u >> 3) < 196) val = ((const h8*)(qg + base))[u];
    else for (int e = 0; e < 8; ++e) val[e] = (_Float16)0.f;
    ((h8*)qs)[u] = val;
  }
  for (int u = tid; u < 224 * 8; u += 256) {
    h8 val;
    if ((u >> 3) < 196) val = ((const h8*)(kg + base))[u];
    else for (int e = 0; e < 8; ++e) val[e] = (_Float16)0.f;
    ((h8*)ks)[u] = val;
  }
  for (int u = tid; u < 224 * 64; u += 256) {       // v transposed: vt[d][n]
    int n = u >> 6, d = u & 63;
    vt[d * 224 + n] = (n < 196) ? vg[base + (size_t)n * 64 + d] : (_Float16)0.f;
  }
  for (int u = tid; u < 27 * 64; u += 256) { rh[u] = rel_h[u]; rw[u] = rel_w[u]; }
  for (int u = tid; u < 12 * 224; u += 256) S[196 * 224 + u] = 0.f;  // pad rows
  __syncthreads();

  // Phase 1: S = scale * q @ k^T  (13 x 14 WMMA tile jobs over 8 waves)
  const float scale = 0.125f;   // 64^-0.5
  for (int job = wave; job < 182; job += 8) {
    int mt = job / 14, nt = job - mt * 14;
    f8 acc; for (int r = 0; r < 8; ++r) acc[r] = 0.f;
    for (int kt = 0; kt < 2; ++kt) {
      h16 a = ld_afrag(qs, 64, mt * 16, kt * 32, lane);
      h16 b = ld_bfrag(ks, 64, nt * 16, kt * 32, lane);
      acc = WMMA_F16(a, b, acc);
    }
    int mrow = mt * 16 + 8 * (lane >> 4);
    int ncol = nt * 16 + (lane & 15);
    for (int r = 0; r < 8; ++r) {
      int m = mrow + r;
      if (m < 196) S[m * 224 + ncol] = acc[r] * scale;
    }
  }
  __syncthreads();

  // Phase 2a: + bias_h  (query (i,j) x key-row i': dot(q, rel_h[i-i'+13]))
  for (int p = tid; p < 196 * 14; p += 256) {
    int m = p / 14, i2 = p - m * 14;
    int i0 = m / 14;
    const float* rv = rh + (i0 - i2 + 13) * 64;
    const _Float16* qr = qs + m * 64;
    float dot = 0.f;
    for (int e = 0; e < 64; ++e) dot += (float)qr[e] * rv[e];
    float* sr = S + m * 224 + i2 * 14;
    for (int j2 = 0; j2 < 14; ++j2) sr[j2] += dot;
  }
  __syncthreads();
  // Phase 2b: + bias_w
  for (int p = tid; p < 196 * 14; p += 256) {
    int m = p / 14, j2 = p - m * 14;
    int j0 = m % 14;
    const float* rv = rw + (j0 - j2 + 13) * 64;
    const _Float16* qr = qs + m * 64;
    float dot = 0.f;
    for (int e = 0; e < 64; ++e) dot += (float)qr[e] * rv[e];
    for (int i2 = 0; i2 < 14; ++i2) S[m * 224 + i2 * 14 + j2] += dot;
  }
  __syncthreads();

  // Phase 3: softmax, one wave per row
  for (int m = wave; m < 196; m += 8) {
    float* row = S + m * 224;
    float mx = -3.0e38f;
    for (int c = lane; c < 196; c += 32) mx = fmaxf(mx, row[c]);
    for (int off = 16; off > 0; off >>= 1) mx = fmaxf(mx, __shfl_xor(mx, off, 32));
    float sum = 0.f;
    for (int c = lane; c < 224; c += 32) {
      float e = (c < 196) ? __expf(row[c] - mx) : 0.f;
      row[c] = e; sum += e;
    }
    for (int off = 16; off > 0; off >>= 1) sum += __shfl_xor(sum, off, 32);
    float inv = 1.f / sum;
    for (int c = lane; c < 224; c += 32) row[c] *= inv;
  }
  __syncthreads();

  // Phase 4: O = P @ V  (13 x 4 tile jobs, K=224)
  for (int job = wave; job < 52; job += 8) {
    int mt = job >> 2, jt = job & 3;
    f8 acc; for (int r = 0; r < 8; ++r) acc[r] = 0.f;
    for (int kt = 0; kt < 7; ++kt) {
      h16 a;                                    // f32 -> f16 on the fly
      int row = mt * 16 + (lane & 15);
      int g = lane >> 4;
      const float* sr = S + row * 224 + kt * 32;
      for (int e = 0; e < 16; ++e) {
        int ko = (e & 7) + 8 * g + ((e >> 3) << 4);
        a[e] = (_Float16)sr[ko];
      }
      h16 b = ld_bfrag(vt, 224, jt * 16, kt * 32, lane);
      acc = WMMA_F16(a, b, acc);
    }
    int mrow = mt * 16 + 8 * (lane >> 4);
    int d = jt * 16 + (lane & 15);
    for (int r = 0; r < 8; ++r) {
      int m = mrow + r;
      if (m < 196)
        o[((size_t)bw * 196 + m) * 768 + head * 64 + d] = (_Float16)acc[r];
    }
  }
}

// ---------------------------------------------------------------------------
extern "C" void kernel_launch(void* const* d_in, const int* in_sizes, int n_in,
                              void* d_out, int out_size, void* d_ws, size_t ws_size,
                              hipStream_t stream) {
  const float* x      = (const float*)d_in[0];
  const float* ln1_g  = (const float*)d_in[1];
  const float* ln1_b  = (const float*)d_in[2];
  const float* qkv_w  = (const float*)d_in[3];
  const float* qkv_b  = (const float*)d_in[4];
  const float* proj_w = (const float*)d_in[5];
  const float* proj_b = (const float*)d_in[6];
  const float* rel_h  = (const float*)d_in[7];
  const float* rel_w  = (const float*)d_in[8];
  const float* ln2_g  = (const float*)d_in[9];
  const float* ln2_b  = (const float*)d_in[10];
  const float* mlp_w1 = (const float*)d_in[11];
  const float* mlp_b1 = (const float*)d_in[12];
  const float* mlp_w2 = (const float*)d_in[13];
  const float* mlp_b2 = (const float*)d_in[14];
  float* out = (float*)d_out;

  char* ws = (char*)d_ws;
  size_t off = 0;
  auto alloc = [&](size_t bytes) {
    size_t o = off; off += (bytes + 255) & ~(size_t)255; return o;
  };
  _Float16* wqkv_t  = (_Float16*)(ws + alloc(2304ULL * 768 * 2));
  _Float16* wproj_t = (_Float16*)(ws + alloc(768ULL * 768 * 2));
  _Float16* w1_t    = (_Float16*)(ws + alloc(3072ULL * 768 * 2));
  _Float16* w2_t    = (_Float16*)(ws + alloc(768ULL * 3072 * 2));
  _Float16* xwin    = (_Float16*)(ws + alloc(39200ULL * 768 * 2));  // reused: h2
  size_t qkvRegion = off;
  _Float16* q = (_Float16*)(ws + alloc(2400ULL * 196 * 64 * 2));
  _Float16* k = (_Float16*)(ws + alloc(2400ULL * 196 * 64 * 2));
  _Float16* v = (_Float16*)(ws + alloc(2400ULL * 196 * 64 * 2));
  _Float16* o = (_Float16*)(ws + alloc(39200ULL * 768 * 2));
  _Float16* mid = (_Float16*)(ws + qkvRegion);   // 32768x3072 f16 overlays q/k/v/o
  _Float16* h2  = xwin;

  dim3 tb(32, 8);
  // weights -> f16 [N][K]
  k_wt<<<dim3(2304 / 32, 768 / 32), tb, 0, stream>>>(qkv_w,  wqkv_t,  768, 2304);
  k_wt<<<dim3(768 / 32,  768 / 32), tb, 0, stream>>>(proj_w, wproj_t, 768, 768);
  k_wt<<<dim3(3072 / 32, 768 / 32), tb, 0, stream>>>(mlp_w1, w1_t,    768, 3072);
  k_wt<<<dim3(768 / 32, 3072 / 32), tb, 0, stream>>>(mlp_w2, w2_t,   3072, 768);

  // LN1 + window partition (pad -> zero tokens)
  k_ln1<<<39200, 256, 0, stream>>>(x, ln1_g, ln1_b, xwin);

  // QKV GEMM (39200 x 2304, K=768) -> head-major q/k/v
  k_gemm_qkv<<<dim3(307, 18), 256, 0, stream>>>(xwin, wqkv_t, qkv_b, q, k, v,
                                                39200, 768);
  // Attention per (window, head)
  k_attn<<<2400, 256, SM_TOT, stream>>>(q, k, v, rel_h, rel_w, o);

  // proj GEMM + unpartition + crop + residual -> x2 in d_out
  k_gemm_proj<<<dim3(307, 6), 256, 0, stream>>>(o, wproj_t, proj_b, x, out,
                                                39200, 768);
  // LN2 on x2
  k_ln2<<<32768, 256, 0, stream>>>(out, ln2_g, ln2_b, h2);

  // MLP
  k_gemm_gelu<<<dim3(256, 24), 256, 0, stream>>>(h2, w1_t, mlp_b1, mid,
                                                 32768, 768);
  k_gemm_out<<<dim3(256, 6), 256, 0, stream>>>(mid, w2_t, mlp_b2, out,
                                               32768, 3072);
}
